// EfficientGraphConv_48696339202115
// MI455X (gfx1250) — matrix operand order repaired and checked by reading, not compiled, not run on hardware
//
#include <hip/hip_runtime.h>
#include <hip/hip_bf16.h>

// ---------------------------------------------------------------------------
// EfficientGraphConv for MI455X (gfx1250, wave32, WMMA)
//   phase 1: node MLP  x[100000,64] -> xt[100000,64]      (WMMA f32<=f16)
//   phase 2: edge MLP  edge_attr[1M,32] -> e_t, fused gather*e_t*dp -> atomic
//            scatter-add into agg[100000,64]
//   phase 3: out = LayerNorm(x + agg*res_w)*gamma + beta
// ---------------------------------------------------------------------------

typedef __attribute__((ext_vector_type(16))) _Float16 v16h;
typedef __attribute__((ext_vector_type(8)))  _Float16 v8h;
typedef __attribute__((ext_vector_type(8)))  float    v8f;

#define N_NODES 100000
#define N_EDGES 1000000

static __device__ __forceinline__ v8f wmma_f16(v16h a, v16h b, v8f c) {
    // D = A(16x32 f16) * B(32x16 f16) + C(16x16 f32)
    return __builtin_amdgcn_wmma_f32_16x16x32_f16(
        /*neg_a=*/false, a, /*neg_b=*/false, b,
        /*c_mod=*/(short)0, c, /*reuse_a=*/false, /*reuse_b=*/false);
}

static __device__ __forceinline__ v8f splat8(float f) {
    v8f r;
#pragma unroll
    for (int i = 0; i < 8; ++i) r[i] = f;
    return r;
}

static __device__ __forceinline__ v16h cat8(v8h lo, v8h hi) {
    return __builtin_shufflevector(lo, hi, 0, 1, 2, 3, 4, 5, 6, 7,
                                   8, 9, 10, 11, 12, 13, 14, 15);
}

// 8 consecutive f32 -> 8 f16 (two b128 global loads + cvt)
static __device__ __forceinline__ v8h cvt8(const float* __restrict__ p) {
    const float4* q = reinterpret_cast<const float4*>(p);
    float4 a = q[0], b = q[1];
    v8h r;
    r[0] = (_Float16)a.x; r[1] = (_Float16)a.y;
    r[2] = (_Float16)a.z; r[3] = (_Float16)a.w;
    r[4] = (_Float16)b.x; r[5] = (_Float16)b.y;
    r[6] = (_Float16)b.z; r[7] = (_Float16)b.w;
    return r;
}

// A fragment (16x32, lane=M row, K chunks at kc and kc+16) from f32 row
static __device__ __forceinline__ v16h load_a_global(const float* __restrict__ row,
                                                     int kc) {
    return cat8(cvt8(row + kc), cvt8(row + kc + 16));
}

// A fragment from f16 LDS row (stride halfs), chunk base column kc
static __device__ __forceinline__ v16h load_a_lds(const _Float16* h, int kc) {
    v8h lo = *(const v8h*)(h + kc);
    v8h hi = *(const v8h*)(h + kc + 16);
    return cat8(lo, hi);
}

// B fragment (32x16): lane holds N col = ncol + lane%16, K at 8h+i and 16+8h+i of the step
static __device__ __forceinline__ v16h load_b_lds(const _Float16* w, int row,
                                                  int ncol, int stride) {
    int kb = (row & ~31) + ((row >> 4) & 1) * 8;
    const _Float16* p = w + ncol + (row & 15);
    v16h r;
#pragma unroll
    for (int i = 0; i < 8; ++i) {
        r[i] = p[(kb + i) * stride];
        r[8 + i] = p[(kb + 16 + i) * stride];
    }
    return r;
}

// ---------------------------------------------------------------------------
// phase 0: zero the aggregation buffer
// ---------------------------------------------------------------------------
__global__ void __launch_bounds__(256) egc_zero_kernel(float4* __restrict__ p, int n4) {
    int i = blockIdx.x * 256 + threadIdx.x;
    if (i < n4) p[i] = make_float4(0.f, 0.f, 0.f, 0.f);
}

// ---------------------------------------------------------------------------
// phase 1: node MLP   xt = (leaky(x@W1+b1))@W2 + b2
//   W1:[64,128]  W2:[128,64]   4 waves / block, one 16-row tile per wave
// ---------------------------------------------------------------------------
__global__ void __launch_bounds__(128)
egc_node_mlp_kernel(const float* __restrict__ x,
                    const float* __restrict__ W1, const float* __restrict__ b1,
                    const float* __restrict__ W2, const float* __restrict__ b2,
                    float* __restrict__ xt) {
    __shared__ _Float16 sW1[64 * 128];   // 16 KB
    __shared__ _Float16 sW2[128 * 64];   // 16 KB
    __shared__ _Float16 sH[4][16 * 128]; // 16 KB  (per-wave hidden tile)

    int tid = threadIdx.x;
    for (int i = tid; i < 64 * 128; i += 128) sW1[i] = (_Float16)W1[i];
    for (int i = tid; i < 128 * 64; i += 128) sW2[i] = (_Float16)W2[i];
    __syncthreads();

    int wave = tid >> 5, lane = tid & 31;
    int tile = blockIdx.x * 4 + wave;               // wave-uniform guard
    if (tile >= N_NODES / 16) return;
    int R = tile * 16;
    int mrow = lane & 15, khalf = lane >> 4, kbase = khalf * 8;

    const float* xrow = x + (size_t)(R + mrow) * 64;
    v16h a0 = load_a_global(xrow, kbase);        // K 0..31
    v16h a1 = load_a_global(xrow, 32 + kbase);   // K 32..63

    _Float16* h = sH[wave];

    // layer 1: 8 N-tiles of 16, K=64 (2 WMMAs each), leaky -> f16 LDS
#pragma unroll
    for (int nt = 0; nt < 8; ++nt) {
        v8f c = splat8(b1[nt * 16 + mrow]);
        v16h bf0 = load_b_lds(sW1, lane, nt * 16, 128);
        v16h bf1 = load_b_lds(sW1, 32 + lane, nt * 16, 128);
        c = wmma_f16(a0, bf0, c);
        c = wmma_f16(a1, bf1, c);
        int col = nt * 16 + mrow;
#pragma unroll
        for (int v = 0; v < 8; ++v) {
            float t = c[v];
            t = fmaxf(t, 0.1f * t);                         // LeakyReLU(0.1)
            h[(v + khalf * 8) * 128 + col] = (_Float16)t;   // D-layout scatter
        }
    }

    // layer 2: A frags from LDS hidden (K=128), 4 N-tiles of 16
    v16h ha[4];
#pragma unroll
    for (int kt = 0; kt < 4; ++kt)
        ha[kt] = load_a_lds(h + mrow * 128, kt * 32 + kbase);

#pragma unroll
    for (int nt = 0; nt < 4; ++nt) {
        v8f c = splat8(b2[nt * 16 + mrow]);
#pragma unroll
        for (int kt = 0; kt < 4; ++kt) {
            v16h bf = load_b_lds(sW2, kt * 32 + lane, nt * 16, 64);
            c = wmma_f16(ha[kt], bf, c);
        }
#pragma unroll
        for (int v = 0; v < 8; ++v)
            xt[(size_t)(R + v + khalf * 8) * 64 + nt * 16 + mrow] = c[v];
    }
}

// ---------------------------------------------------------------------------
// phase 2: edge MLP + fused gather-multiply-scatter
//   e_t = (leaky(edge_attr@W1e+b1e))@W2e + b2e   [16 edges / wave]
//   agg[dst] += xt[src] * e_t * dp_scale   (global f32 atomics, L2-resident)
// ---------------------------------------------------------------------------
__global__ void __launch_bounds__(256)
egc_edge_kernel(const float* __restrict__ ea,
                const float* __restrict__ W1, const float* __restrict__ b1,
                const float* __restrict__ W2, const float* __restrict__ b2,
                const float* __restrict__ xt,
                const int* __restrict__ src, const int* __restrict__ dst,
                const float* __restrict__ dp_scale,
                float* __restrict__ agg) {
    __shared__ _Float16 sW1[32 * 64];   // 4 KB
    __shared__ _Float16 sW2[64 * 64];   // 8 KB
    __shared__ _Float16 sH[8][16 * 64]; // 16 KB

    int tid = threadIdx.x;
    for (int i = tid; i < 32 * 64; i += 256) sW1[i] = (_Float16)W1[i];
    for (int i = tid; i < 64 * 64; i += 256) sW2[i] = (_Float16)W2[i];
    __syncthreads();

    int wave = tid >> 5, lane = tid & 31;
    int tile = blockIdx.x * 8 + wave;               // wave-uniform guard
    if (tile >= N_EDGES / 16) return;
    int E0 = tile * 16;
    int mrow = lane & 15, khalf = lane >> 4, kbase = khalf * 8;

    // layer 1: single A frag (K=32)
    const float* erow = ea + (size_t)(E0 + mrow) * 32;
    v16h a = load_a_global(erow, kbase);

    _Float16* h = sH[wave];
#pragma unroll
    for (int nt = 0; nt < 4; ++nt) {
        v8f c = splat8(b1[nt * 16 + mrow]);
        v16h bf = load_b_lds(sW1, lane, nt * 16, 64);
        c = wmma_f16(a, bf, c);
        int col = nt * 16 + mrow;
#pragma unroll
        for (int v = 0; v < 8; ++v) {
            float t = c[v];
            t = fmaxf(t, 0.1f * t);
            h[(v + khalf * 8) * 64 + col] = (_Float16)t;
        }
    }

    // layer 2: K=64, keep all 4 output tiles live for the scatter
    v16h ha0 = load_a_lds(h + mrow * 64, kbase);
    v16h ha1 = load_a_lds(h + mrow * 64, 32 + kbase);

    v8f acc[4];
#pragma unroll
    for (int nt = 0; nt < 4; ++nt) {
        v8f c = splat8(b2[nt * 16 + mrow]);
        c = wmma_f16(ha0, load_b_lds(sW2, lane, nt * 16, 64), c);
        c = wmma_f16(ha1, load_b_lds(sW2, 32 + lane, nt * 16, 64), c);
        acc[nt] = c;
    }

    float dp = dp_scale[0];
    // D layout: lane holds column (nt*16+mrow) of edge rows M = v + 8*khalf
#pragma unroll
    for (int v = 0; v < 8; ++v) {
        int e = E0 + v + khalf * 8;
        int s = src[e];
        int d = dst[e];
        const float* xs = xt + (size_t)s * 64;
        float* ad = agg + (size_t)d * 64;
#pragma unroll
        for (int nt = 0; nt < 4; ++nt) {
            int col = nt * 16 + mrow;
            float val = xs[col] * acc[nt][v] * dp;
            atomicAdd(&ad[col], val);
        }
    }
}

// ---------------------------------------------------------------------------
// phase 3: out = LayerNorm(x + agg*res_w); one wave per row, 2 cols/lane
// ---------------------------------------------------------------------------
__global__ void __launch_bounds__(256)
egc_final_kernel(const float* __restrict__ x, const float* __restrict__ agg,
                 const float* __restrict__ res_w,
                 const float* __restrict__ gamma, const float* __restrict__ beta,
                 float* __restrict__ out) {
    int wave = threadIdx.x >> 5, lane = threadIdx.x & 31;
    int row = blockIdx.x * 8 + wave;
    if (row >= N_NODES) return;
    float rw = res_w[0];
    size_t base = (size_t)row * 64;
    float v0 = x[base + lane] + agg[base + lane] * rw;
    float v1 = x[base + lane + 32] + agg[base + lane + 32] * rw;
    float s = v0 + v1;
    float q = v0 * v0 + v1 * v1;
#pragma unroll
    for (int off = 16; off > 0; off >>= 1) {
        s += __shfl_xor(s, off, 32);
        q += __shfl_xor(q, off, 32);
    }
    float mu = s * (1.0f / 64.0f);
    float var = q * (1.0f / 64.0f) - mu * mu;
    float rs = rsqrtf(var + 1e-5f);
    out[base + lane]      = (v0 - mu) * rs * gamma[lane]      + beta[lane];
    out[base + lane + 32] = (v1 - mu) * rs * gamma[lane + 32] + beta[lane + 32];
}

// ---------------------------------------------------------------------------
extern "C" void kernel_launch(void* const* d_in, const int* in_sizes, int n_in,
                              void* d_out, int out_size, void* d_ws, size_t ws_size,
                              hipStream_t stream) {
    const float* x    = (const float*)d_in[0];
    const float* ea   = (const float*)d_in[1];
    const float* W1n  = (const float*)d_in[2];
    const float* b1n  = (const float*)d_in[3];
    const float* W2n  = (const float*)d_in[4];
    const float* b2n  = (const float*)d_in[5];
    const float* W1e  = (const float*)d_in[6];
    const float* b1e  = (const float*)d_in[7];
    const float* W2e  = (const float*)d_in[8];
    const float* b2e  = (const float*)d_in[9];
    const float* resw = (const float*)d_in[10];
    const float* dps  = (const float*)d_in[11];
    const float* gam  = (const float*)d_in[12];
    const float* bet  = (const float*)d_in[13];
    const int*   ei   = (const int*)d_in[14];

    float* xt  = (float*)d_ws;                       // [100000,64] f32
    float* agg = xt + (size_t)N_NODES * 64;          // [100000,64] f32
    float* out = (float*)d_out;

    const int n4 = N_NODES * 64 / 4;                 // 1.6M float4
    egc_zero_kernel<<<(n4 + 255) / 256, 256, 0, stream>>>((float4*)agg, n4);

    // 6250 node tiles / 4 waves per block
    egc_node_mlp_kernel<<<(N_NODES / 16 + 3) / 4, 128, 0, stream>>>(
        x, W1n, b1n, W2n, b2n, xt);

    // 62500 edge tiles / 8 waves per block
    egc_edge_kernel<<<(N_EDGES / 16 + 7) / 8, 256, 0, stream>>>(
        ea, W1e, b1e, W2e, b2e, xt, ei, ei + N_EDGES, dps, agg);

    // 100000 rows / 8 waves per block
    egc_final_kernel<<<N_NODES / 8, 256, 0, stream>>>(x, agg, resw, gam, bet, out);
}
